// GNNEncoder_70566312673936
// MI455X (gfx1250) — matrix operand order repaired and checked by reading, not compiled, not run on hardware
//
#include <hip/hip_runtime.h>

// ---------------------------------------------------------------------------
// CDNA5 / gfx1250 bipartite GNN encoder.
// All 64/128-deep matmuls on v_wmma_f32_16x16x32_bf16 (wave32), fp32 accum.
// Weights pre-packed to bf16 fragment-major (2x global_load_b128 / fragment);
// LDS activation tiles stored as packed bf16 (2x ds_load_b128 / A fragment).
// Edge path fully fused: gather + LN + WMMA + atomic scatter.
// din=1 edge projection folded exactly to a per-conv constant vector.
// ---------------------------------------------------------------------------

typedef __attribute__((ext_vector_type(16))) __bf16 v16bf;
typedef __attribute__((ext_vector_type(8)))  __bf16 v8bf;
typedef __attribute__((ext_vector_type(8)))  float  v8f;

#define LN_EPS 1e-5f

__device__ __forceinline__ unsigned pack_bf16(float lo, float hi) {
  unsigned short a = __builtin_bit_cast(unsigned short, (__bf16)lo);
  unsigned short b = __builtin_bit_cast(unsigned short, (__bf16)hi);
  return ((unsigned)b << 16) | a;  // lowers to v_cvt_pk_bf16_f32
}

// ---- WMMA fragment loaders (CDNA5 ISA 7.12.2 layouts, wave32) --------------
// A: 16x32 bf16 from bf16 LDS tile (row-major, lda elements).
// lanes 0-15: row m=lane, k in {0..7,16..23}; lanes 16-31: k in {8..15,24..31}.
__device__ __forceinline__ v16bf load_a_bf(const __bf16* a, int lda, int k0) {
  int lane = threadIdx.x & 31;
  int m    = lane & 15;
  int klo  = (lane & 16) ? 8 : 0;
  const __bf16* p = a + m * lda + k0 + klo;
  v8bf lo = *(const v8bf*)p;         // ds_load_b128
  v8bf hi = *(const v8bf*)(p + 16);  // ds_load_b128
  return __builtin_shufflevector(lo, hi, 0, 1, 2, 3, 4, 5, 6, 7,
                                 8, 9, 10, 11, 12, 13, 14, 15);
}

// B: packed fragment-major weights; lane's 16 bf16 contiguous.
__device__ __forceinline__ v16bf load_b_packed(const __bf16* wp, int frag) {
  int lane = threadIdx.x & 31;
  return *(const v16bf*)(wp + ((long long)(frag * 32 + lane)) * 16);
}

__device__ __forceinline__ void zero_acc(v8f acc[4]) {
#pragma unroll
  for (int nt = 0; nt < 4; ++nt)
#pragma unroll
    for (int i = 0; i < 8; ++i) acc[nt][i] = 0.f;
}

// (16 x K) @ (K x 64); A from bf16 LDS, W packed bf16.
template <int K>
__device__ __forceinline__ void mm_bf(const __bf16* a, int lda,
                                      const __bf16* wp, v8f acc[4]) {
#pragma unroll
  for (int kt = 0; kt < K / 32; ++kt) {
    v16bf af = load_a_bf(a, lda, kt * 32);
#pragma unroll
    for (int nt = 0; nt < 4; ++nt) {
      v16bf bf = load_b_packed(wp, kt * 4 + nt);
      acc[nt] = __builtin_amdgcn_wmma_f32_16x16x32_bf16(
          false, af, false, bf, (short)0, acc[nt], false, false);
    }
  }
}

// C/D layout: lane -> col n=lane&15; vgpr r -> row m = r + (lane<16?0:8).
__device__ __forceinline__ void store_tile_f32(const v8f acc[4], const float* bias,
                                               int relu, float* out, int ldo,
                                               int rows_valid) {
  int lane = threadIdx.x & 31;
  int n    = lane & 15;
  int mb   = (lane & 16) ? 8 : 0;
#pragma unroll
  for (int nt = 0; nt < 4; ++nt) {
    float b = bias ? bias[nt * 16 + n] : 0.f;
#pragma unroll
    for (int r = 0; r < 8; ++r) {
      int m = mb + r;
      if (m < rows_valid) {
        float v = acc[nt][r] + b;
        if (relu) v = fmaxf(v, 0.f);
        out[(long long)m * ldo + nt * 16 + n] = v;
      }
    }
  }
}

__device__ __forceinline__ void store_tile_lds_bf16(const v8f acc[4], const float* bias,
                                                    __bf16* out, int ldo) {
  int lane = threadIdx.x & 31;
  int n    = lane & 15;
  int mb   = (lane & 16) ? 8 : 0;
#pragma unroll
  for (int nt = 0; nt < 4; ++nt) {
    float b = bias[nt * 16 + n];
#pragma unroll
    for (int r = 0; r < 8; ++r)
      out[(mb + r) * ldo + nt * 16 + n] = (__bf16)fmaxf(acc[nt][r] + b, 0.f);
  }
}

// Coalesced stage of 128 rows x 64 f32 -> packed bf16 LDS tile (row clamp).
__device__ __forceinline__ void stage_rows_bf16(const float* src, long long row0,
                                                long long nrows, __bf16* dst) {
  int t = threadIdx.x;  // 256 threads, 32 floats each
  unsigned* du = (unsigned*)dst;
#pragma unroll
  for (int i = 0; i < 32; i += 4) {
    int idx = t * 32 + i;
    long long r = row0 + idx / 64;
    int c = idx & 63;
    long long rc = (r < nrows) ? r : (nrows - 1);
    float4 v = *(const float4*)(src + rc * 64 + c);   // global_load_b128
    uint2 p;
    p.x = pack_bf16(v.x, v.y);
    p.y = pack_bf16(v.z, v.w);
    *(uint2*)(du + idx / 2) = p;                      // ds_store_b64
  }
}

// ---------------------------------------------------------------------------
__global__ void zero_kernel(float* __restrict__ p, long long n) {
  long long i = (long long)blockIdx.x * blockDim.x + threadIdx.x;
  long long st = (long long)gridDim.x * blockDim.x;
  for (; i < n; i += st) p[i] = 0.f;
}

// Pack f32 W[K x 64] -> bf16 fragment-major: out[((kt*4+nt)*32+lane)*16+i]
// holds W[kt*32 + (lane&16?8:0) + (i<8?i:i+8)][nt*16 + (lane&15)].
__global__ void pack_w_kernel(const float* __restrict__ W, int total,
                              __bf16* __restrict__ out) {
  int idx = blockIdx.x * 256 + threadIdx.x;
  if (idx >= total) return;
  int i    = idx & 15;
  int lane = (idx >> 4) & 31;
  int frag = idx >> 9;
  int kt   = frag >> 2;
  int nt   = frag & 3;
  int k = kt * 32 + ((lane & 16) ? 8 : 0) + (i < 8 ? i : i + 8);
  int n = nt * 16 + (lane & 15);
  out[idx] = (__bf16)W[k * 64 + n];
}

// out[N,64] = opt_relu(in[N,64] @ W + bias), W packed bf16.
__global__ void __launch_bounds__(256)
mm64_kernel(const float* __restrict__ in, int nrows,
            const __bf16* __restrict__ Wp, const float* __restrict__ bias,
            int relu, float* __restrict__ out) {
  __shared__ __bf16 smA[128 * 64];
  long long row0 = (long long)blockIdx.x * 128;
  stage_rows_bf16(in, row0, nrows, smA);
  __syncthreads();
  int wave = threadIdx.x >> 5;
  long long wr0 = row0 + wave * 16;
  v8f acc[4];
  zero_acc(acc);
  mm_bf<64>(smA + wave * 16 * 64, 64, Wp, acc);
  long long rem = (long long)nrows - wr0;
  int rv = rem > 16 ? 16 : (rem < 0 ? 0 : (int)rem);
  store_tile_f32(acc, bias, relu, out + wr0 * 64, 64, rv);
}

// LN(din) -> relu(x@w1+b1) -> relu(x1@w2+b2); w2 packed bf16. 128 rows/block.
template <int DIN>
__global__ void __launch_bounds__(256)
proj_kernel(const float* __restrict__ x, int nrows,
            const float* __restrict__ g, const float* __restrict__ b,
            const float* __restrict__ w1, const float* __restrict__ b1,
            const __bf16* __restrict__ w2p, const float* __restrict__ b2,
            float* __restrict__ out) {
  __shared__ __bf16 smX[128 * 64];
  long long row0 = (long long)blockIdx.x * 128;
  int t = threadIdx.x;
  int lr = t >> 1, hf = t & 1;
  long long r = row0 + lr;
  long long rc = (r < nrows) ? r : (nrows - 1);
  float xv[DIN];
  float m = 0.f;
#pragma unroll
  for (int i = 0; i < DIN; ++i) { xv[i] = x[rc * DIN + i]; m += xv[i]; }
  m *= (1.f / DIN);
  float var = 0.f;
#pragma unroll
  for (int i = 0; i < DIN; ++i) { float d = xv[i] - m; var += d * d; }
  var *= (1.f / DIN);
  float rs = rsqrtf(var + LN_EPS);
#pragma unroll
  for (int i = 0; i < DIN; ++i) xv[i] = (xv[i] - m) * rs * g[i] + b[i];
  unsigned* smXu = (unsigned*)smX;
  for (int j = 0; j < 16; ++j) {
    int c0 = hf * 32 + 2 * j;
    float s0 = b1[c0], s1 = b1[c0 + 1];
#pragma unroll
    for (int i = 0; i < DIN; ++i) {
      s0 += xv[i] * w1[i * 64 + c0];
      s1 += xv[i] * w1[i * 64 + c0 + 1];
    }
    smXu[lr * 32 + hf * 16 + j] = pack_bf16(fmaxf(s0, 0.f), fmaxf(s1, 0.f));
  }
  __syncthreads();
  int wave = t >> 5;
  long long wr0 = row0 + wave * 16;
  v8f acc[4];
  zero_acc(acc);
  mm_bf<64>(smX + wave * 16 * 64, 64, w2p, acc);
  long long rem = (long long)nrows - wr0;
  int rv = rem > 16 ? 16 : (rem < 0 ? 0 : (int)rem);
  store_tile_f32(acc, b2, 1, out + wr0 * 64, 64, rv);
}

// din=1 edge projection folds exactly: LN(x) == ln_b; e1 and e1@we are
// data-independent 64-vectors (one per conv).
__global__ void econst_kernel(const float* __restrict__ elnb,
                              const float* __restrict__ ew1, const float* __restrict__ eb1,
                              const float* __restrict__ ew2, const float* __restrict__ eb2,
                              const float* __restrict__ we0, const float* __restrict__ we1,
                              const float* __restrict__ we2, const float* __restrict__ we3,
                              float* __restrict__ e1out, float* __restrict__ eweout) {
  __shared__ float x1[64], e1[64];
  int c = threadIdx.x;
  float xn = elnb[0];                        // exact: (x-mean)==0 for din=1
  x1[c] = fmaxf(xn * ew1[c] + eb1[c], 0.f);  // w1 is (1,64)
  __syncthreads();
  float a = eb2[c];
  for (int k = 0; k < 64; ++k) a += x1[k] * ew2[k * 64 + c];
  a = fmaxf(a, 0.f);
  e1[c] = a;
  e1out[c] = a;
  __syncthreads();
  const float* wes[4] = {we0, we1, we2, we3};
#pragma unroll
  for (int j = 0; j < 4; ++j) {
    float s = 0.f;
    for (int k = 0; k < 64; ++k) s += e1[k] * wes[j][k * 64 + c];
    eweout[j * 64 + c] = s;
  }
}

// Fused edge kernel: 128 edges/block (256 thr).
// m = tl[dst] + sr[src] + ewe_const ; msg = relu(LN(m)) @ fw + fb ;
// atomicAdd into agg[dst] directly from the WMMA C layout.
__global__ void __launch_bounds__(256)
edge_kernel(const int* __restrict__ dst, const int* __restrict__ src,
            long long nedges,
            const float* __restrict__ tl, const float* __restrict__ sr,
            const float* __restrict__ ewe,
            const float* __restrict__ flng, const float* __restrict__ flnb,
            const __bf16* __restrict__ fwp, const float* __restrict__ fb,
            float* __restrict__ agg) {
  __shared__ __bf16 smM[128 * 64];
  __shared__ int smD[128];
  __shared__ int smS[128];
  long long e0 = (long long)blockIdx.x * 128;
  int t = threadIdx.x;
  if (t < 128) {
    long long e = e0 + t;
    if (e >= nedges) e = nedges - 1;
    smD[t] = dst[e];
    smS[t] = src[e];
  }
  __syncthreads();

  int lr = t >> 1, hf = t & 1;
  const float* trow = tl + (long long)smD[lr] * 64 + hf * 32;
  const float* srow = sr + (long long)smS[lr] * 64 + hf * 32;
  const float* erow = ewe + hf * 32;
  float v[32];
  float sum = 0.f, ssq = 0.f;
#pragma unroll
  for (int i = 0; i < 32; ++i) {
    float x = trow[i] + srow[i] + erow[i];
    v[i] = x; sum += x; ssq += x * x;
  }
  sum += __shfl_xor(sum, 1, 32);
  ssq += __shfl_xor(ssq, 1, 32);
  float mean = sum * (1.f / 64.f);
  float var  = ssq * (1.f / 64.f) - mean * mean;
  float rs   = rsqrtf(var + LN_EPS);
  unsigned* smMu = (unsigned*)smM;
#pragma unroll
  for (int j = 0; j < 16; ++j) {
    int c0 = hf * 32 + 2 * j;
    float a0 = fmaxf((v[2 * j] - mean) * rs * flng[c0] + flnb[c0], 0.f);
    float a1 = fmaxf((v[2 * j + 1] - mean) * rs * flng[c0 + 1] + flnb[c0 + 1], 0.f);
    smMu[lr * 32 + hf * 16 + j] = pack_bf16(a0, a1);
  }
  __syncthreads();

  int wave = t >> 5;
  long long we0 = e0 + wave * 16;
  v8f acc[4];
  zero_acc(acc);
  mm_bf<64>(smM + wave * 16 * 64, 64, fwp, acc);
  int lane = t & 31;
  int n = lane & 15;
  int mb = (lane & 16) ? 8 : 0;
#pragma unroll
  for (int nt = 0; nt < 4; ++nt) {
    float bb = fb[nt * 16 + n];
#pragma unroll
    for (int r = 0; r < 8; ++r) {
      int m = mb + r;
      long long ee = we0 + m;
      if (ee < nedges) {
        int dd = smD[wave * 16 + m];
        atomicAdd(agg + (long long)dd * 64 + nt * 16 + n, acc[nt][r] + bb);
      }
    }
  }
}

// Fused node output: h = [LN(agg) , right] (128) -> relu(h@ow1+ob1)@ow2+ob2.
// 64 rows/block, 128 threads (4 waves). ow1 (K=128) and ow2 packed bf16.
__global__ void __launch_bounds__(128)
node_out_kernel(const float* __restrict__ agg, const float* __restrict__ right,
                int nrows,
                const float* __restrict__ plng, const float* __restrict__ plnb,
                const __bf16* __restrict__ ow1p, const float* __restrict__ ob1,
                const __bf16* __restrict__ ow2p, const float* __restrict__ ob2,
                float* __restrict__ out) {
  __shared__ __bf16 smH[64 * 128];
  __shared__ __bf16 smX[64 * 64];
  long long row0 = (long long)blockIdx.x * 64;
  int t = threadIdx.x;
  int lr = t >> 1, hf = t & 1;
  long long r = row0 + lr;
  long long rc = (r < nrows) ? r : (nrows - 1);
  const float* arow = agg + rc * 64 + hf * 32;
  const float* rrow = right + rc * 64 + hf * 32;
  float v[32];
  float sum = 0.f, ssq = 0.f;
#pragma unroll
  for (int i = 0; i < 32; ++i) {
    float x = arow[i];
    v[i] = x; sum += x; ssq += x * x;
  }
  sum += __shfl_xor(sum, 1, 32);
  ssq += __shfl_xor(ssq, 1, 32);
  float mean = sum * (1.f / 64.f);
  float var  = ssq * (1.f / 64.f) - mean * mean;
  float rs   = rsqrtf(var + LN_EPS);
  unsigned* smHu = (unsigned*)smH;  // 64 uints per row (128 bf16)
#pragma unroll
  for (int j = 0; j < 16; ++j) {
    int c0 = hf * 32 + 2 * j;
    float a0 = (v[2 * j] - mean) * rs * plng[c0] + plnb[c0];
    float a1 = (v[2 * j + 1] - mean) * rs * plng[c0 + 1] + plnb[c0 + 1];
    smHu[lr * 64 + hf * 16 + j] = pack_bf16(a0, a1);
    smHu[lr * 64 + 32 + hf * 16 + j] = pack_bf16(rrow[2 * j], rrow[2 * j + 1]);
  }
  __syncthreads();

  int wave = t >> 5;
  long long wr0 = row0 + wave * 16;
  v8f acc[4];
  zero_acc(acc);
  mm_bf<128>(smH + wave * 16 * 128, 128, ow1p, acc);
  store_tile_lds_bf16(acc, ob1, smX + wave * 16 * 64, 64);
  __syncthreads();

  v8f acc2[4];
  zero_acc(acc2);
  mm_bf<64>(smX + wave * 16 * 64, 64, ow2p, acc2);
  long long rem = (long long)nrows - wr0;
  int rv = rem > 16 ? 16 : (rem < 0 ? 0 : (int)rem);
  store_tile_f32(acc2, ob2, 0, out + wr0 * 64, 64, rv);
}

// ---------------------------------------------------------------------------
// Host side
// ---------------------------------------------------------------------------
static void run_conv(hipStream_t stream, void* const* d_in, int base,
                     const float* left, int Nl,
                     const int* srci, const int* dsti, long long nE,
                     const float* right, int Nr,
                     const float* ewe_c, __bf16* const* pk,  // {wl,wr,fw,ow1,ow2}
                     float* tl, float* sr, float* agg, float* out) {
  auto F = [&](int i) { return (const float*)d_in[i]; };
  // conv param order: wl,bl,we,wr,fln_g,fln_b,fw,fb,pln_g,pln_b,ow1,ob1,ow2,ob2
  mm64_kernel<<<(Nr + 127) / 128, 256, 0, stream>>>(right, Nr, pk[0], F(base + 1), 0, tl);
  mm64_kernel<<<(Nl + 127) / 128, 256, 0, stream>>>(left, Nl, pk[1], nullptr, 0, sr);
  zero_kernel<<<512, 256, 0, stream>>>(agg, (long long)Nr * 64);
  edge_kernel<<<(int)((nE + 127) / 128), 256, 0, stream>>>(
      dsti, srci, nE, tl, sr, ewe_c,
      F(base + 4), F(base + 5), pk[2], F(base + 7), agg);
  node_out_kernel<<<(Nr + 63) / 64, 128, 0, stream>>>(
      agg, right, Nr, F(base + 8), F(base + 9),
      pk[3], F(base + 11), pk[4], F(base + 13), out);
}

extern "C" void kernel_launch(void* const* d_in, const int* in_sizes, int n_in,
                              void* d_out, int out_size, void* d_ws, size_t ws_size,
                              hipStream_t stream) {
  (void)n_in; (void)out_size; (void)ws_size;
  const int ncons = in_sizes[0] / 5;
  const long long nedges = in_sizes[1] / 2;
  const int nvars = in_sizes[3] / 19;

  auto F = [&](int i) { return (const float*)d_in[i]; };
  const int* ei = (const int*)d_in[1];
  const int* cons_idx = ei;            // edge_indices[0]
  const int* var_idx  = ei + nedges;   // edge_indices[1]

  float* wsf = (float*)d_ws;
  size_t o = 0;
  auto alloc = [&](size_t n) { float* p = wsf + o; o += n; return p; };
  int nmax = nvars > ncons ? nvars : ncons;
  float* cbuf = alloc((size_t)ncons * 64);
  float* vbuf = alloc((size_t)nvars * 64);
  float* c1   = alloc((size_t)ncons * 64);
  float* v1   = alloc((size_t)nvars * 64);
  float* tl   = alloc((size_t)nmax * 64);
  float* srb  = alloc((size_t)nmax * 64);
  float* agg  = alloc((size_t)nmax * 64);
  float* e1   = alloc(64);
  float* ewe  = alloc(256);

  // bf16 pool (32B aligned: all prior sizes are multiples of 64 floats)
  __bf16* bfp = (__bf16*)(wsf + o);
  size_t ob = 0;
  auto balloc = [&](size_t n) { __bf16* p = bfp + ob; ob += n; return p; };
  auto pack = [&](const float* W, int K, __bf16* dst) {
    int total = K * 64;
    pack_w_kernel<<<(total + 255) / 256, 256, 0, stream>>>(W, total, dst);
  };

  __bf16* pw2c = balloc(64 * 64);
  __bf16* pw2v = balloc(64 * 64);
  __bf16* pconv[4][5];
  const int bases[4] = {22, 36, 50, 64};
  for (int j = 0; j < 4; ++j) {
    pconv[j][0] = balloc(64 * 64);    // wl
    pconv[j][1] = balloc(64 * 64);    // wr
    pconv[j][2] = balloc(64 * 64);    // fw
    pconv[j][3] = balloc(128 * 64);   // ow1
    pconv[j][4] = balloc(64 * 64);    // ow2
  }
  pack(F(8), 64, pw2c);               // cons_proj.w2
  pack(F(14), 64, pw2v);              // var_proj.w2
  for (int j = 0; j < 4; ++j) {
    int b = bases[j];
    pack(F(b + 0), 64, pconv[j][0]);
    pack(F(b + 3), 64, pconv[j][1]);
    pack(F(b + 6), 64, pconv[j][2]);
    pack(F(b + 10), 128, pconv[j][3]);
    pack(F(b + 12), 64, pconv[j][4]);
  }

  // Input projections (cons_proj @ d_in[4..9], var_proj @ [10..15])
  proj_kernel<5><<<(ncons + 127) / 128, 256, 0, stream>>>(
      F(0), ncons, F(4), F(5), F(6), F(7), pw2c, F(9), cbuf);
  proj_kernel<19><<<(nvars + 127) / 128, 256, 0, stream>>>(
      F(3), nvars, F(10), F(11), F(12), F(13), pw2v, F(15), vbuf);
  // Edge projection constant (din=1 LayerNorm == bias); fold e1@we per conv.
  econst_kernel<<<1, 64, 0, stream>>>(
      F(17), F(18), F(19), F(20), F(21),
      F(22 + 2), F(36 + 2), F(50 + 2), F(64 + 2), e1, ewe);

  float* outc = (float*)d_out;
  float* outv = (float*)d_out + (size_t)ncons * 64;

  // c = conv_v_to_c (left=v, src=var_idx, dst=cons_idx, right=c)
  run_conv(stream, d_in, 22, vbuf, nvars, var_idx, cons_idx, nedges,
           cbuf, ncons, ewe + 0, pconv[0], tl, srb, agg, c1);
  // v = conv_c_to_v (left=c, src=cons_idx, dst=var_idx, right=v)
  run_conv(stream, d_in, 36, c1, ncons, cons_idx, var_idx, nedges,
           vbuf, nvars, ewe + 64, pconv[1], tl, srb, agg, v1);
  // c = conv_v_to_c2 -> final c (into d_out)
  run_conv(stream, d_in, 50, v1, nvars, var_idx, cons_idx, nedges,
           c1, ncons, ewe + 128, pconv[2], tl, srb, agg, outc);
  // v = conv_c_to_v2 -> final v (into d_out)
  run_conv(stream, d_in, 64, outc, ncons, cons_idx, var_idx, nedges,
           v1, nvars, ewe + 192, pconv[3], tl, srb, agg, outv);
}
